// MoDInfiniTransformer_68762426409346
// MI455X (gfx1250) — matrix operand order repaired and checked by reading, not compiled, not run on hardware
//
#include <hip/hip_runtime.h>

// ---------------------------------------------------------------------------
// MoD Infini-Transformer layer for MI455X (gfx1250, wave32, WMMA).
// All big GEMMs use v_wmma_f32_16x16x32_bf16 (fp32 accumulate) with
// double-buffered GLOBAL_LOAD_ASYNC_TO_LDS_B128 staging (ASYNCcnt path);
// softmax, linear-memory state, LayerNorm, routing stay in fp32 VALU.
// ---------------------------------------------------------------------------

typedef unsigned short u16;
typedef __attribute__((ext_vector_type(16))) __bf16 v16bf;
typedef __attribute__((ext_vector_type(8)))  float  v8f;

union BF16Frag { v16bf v; unsigned u[8]; };

static __device__ __forceinline__ u16 f2bf(float f) {
    unsigned u = __float_as_uint(f);
    unsigned r = u + 0x7FFFu + ((u >> 16) & 1u);   // RNE
    return (u16)(r >> 16);
}
static __device__ __forceinline__ float bf2f(u16 h) {
    return __uint_as_float(((unsigned)h) << 16);
}
static __device__ __forceinline__ float elu1(float x) {
    return x > 0.f ? x + 1.f : __expf(x);
}
static __device__ __forceinline__ float wave_reduce_add(float v) {
    #pragma unroll
    for (int m = 16; m > 0; m >>= 1) v += __shfl_xor(v, m, 32);
    return v;
}
// low 32 bits of a generic pointer to __shared__ == LDS byte offset
static __device__ __forceinline__ unsigned lds_addr(const void* p) {
    return (unsigned)(unsigned long long)p;
}
// issue two 16B async global->LDS copies (GV mode, at +0 and +16 B); tracked by ASYNCcnt
static __device__ __forceinline__ void async_b128(unsigned lds, const void* gptr) {
    unsigned long long ga = (unsigned long long)gptr;
    asm volatile("global_load_async_to_lds_b128 %0, %1, off\n\t"
                 "global_load_async_to_lds_b128 %0, %1, off offset:16"
                 :: "v"(lds), "v"(ga) : "memory");
}

// Problem constants
#define Bn   4
#define Sn   8192
#define Dn   1024
#define SEGn 512
#define FULLSEG 2048
#define KSEL 512

// ---------------------------------------------------------------------------
// elementwise fp32 -> bf16
__global__ void cvt_kernel(const float* __restrict__ src, u16* __restrict__ dst, int n) {
    int i = (blockIdx.x * 256 + threadIdx.x) * 4;
    if (i < n) {
        dst[i + 0] = f2bf(src[i + 0]);
        dst[i + 1] = f2bf(src[i + 1]);
        dst[i + 2] = f2bf(src[i + 2]);
        dst[i + 3] = f2bf(src[i + 3]);
    }
}

// W (K x N) fp32 -> WT (N x K) bf16
__global__ void tw_kernel(const float* __restrict__ W, u16* __restrict__ WT, int K, int N) {
    int e = blockIdx.x * 256 + threadIdx.x;
    if (e < K * N) {
        int k = e / N, n2 = e - k * N;
        WT[(size_t)n2 * K + k] = f2bf(W[e]);
    }
}

// router scores: one wave per token, D=1024 dot
__global__ void scores_kernel(const float* __restrict__ x, const float* __restrict__ Wps,
                              const float* __restrict__ bps, float* __restrict__ scores) {
    int w = threadIdx.x >> 5, lane = threadIdx.x & 31;
    int tok = blockIdx.x * 8 + w;
    const float4* xv = (const float4*)(x + (size_t)tok * Dn);
    const float4* wv = (const float4*)Wps;
    float s = 0.f;
    for (int i = lane; i < Dn / 4; i += 32) {
        float4 a = xv[i], b = wv[i];
        s += a.x * b.x + a.y * b.y + a.z * b.z + a.w * b.w;
    }
    s = wave_reduce_add(s);
    if (lane == 0) scores[tok] = s + bps[0];
}

// ---------------------------------------------------------------------------
// per (b,seg) top-512 selection: radix-select threshold, stable ties, prefix
// scan in index order (matches jax top_k + sort(idx)).
__global__ void select_kernel(const float* __restrict__ scores,
                              int* __restrict__ flat, int* __restrict__ selpos) {
    int b = blockIdx.x >> 2, seg = blockIdx.x & 3;
    const float* sc = scores + (size_t)b * Sn + seg * FULLSEG;
    __shared__ unsigned keys[FULLSEG];
    __shared__ int part[256];
    __shared__ int scan[256];
    __shared__ int cnt;
    int t = threadIdx.x;
    for (int i = t; i < FULLSEG; i += 256) {
        unsigned u = __float_as_uint(sc[i]);
        keys[i] = (u & 0x80000000u) ? ~u : (u | 0x80000000u);   // order-preserving map
    }
    __syncthreads();
    unsigned prefix = 0;
    for (int bit = 31; bit >= 0; --bit) {
        unsigned cand = prefix | (1u << bit);
        if (t == 0) cnt = 0;
        __syncthreads();
        int lc = 0;
        for (int i = t; i < FULLSEG; i += 256) lc += (keys[i] >= cand) ? 1 : 0;
        atomicAdd(&cnt, lc);
        __syncthreads();
        if (cnt >= KSEL) prefix = cand;   // uniform decision (shared cnt)
        __syncthreads();
    }
    // prefix == K-th largest key T
    if (t == 0) cnt = 0;
    __syncthreads();
    {
        int lc = 0;
        for (int i = t; i < FULLSEG; i += 256) lc += (keys[i] > prefix) ? 1 : 0;
        atomicAdd(&cnt, lc);
    }
    __syncthreads();
    int need = KSEL - cnt;   // ties to keep (earliest indices)
    int i0 = t * 8;
    int sel8[8];
    int ltie = 0;
    #pragma unroll
    for (int j = 0; j < 8; ++j) ltie += (keys[i0 + j] == prefix) ? 1 : 0;
    part[t] = ltie;
    __syncthreads();
    if (t == 0) { int s = 0; for (int i = 0; i < 256; ++i) { scan[i] = s; s += part[i]; } }
    __syncthreads();
    int trank = scan[t];
    int lsel = 0;
    #pragma unroll
    for (int j = 0; j < 8; ++j) {
        int above = keys[i0 + j] > prefix;
        int tie   = keys[i0 + j] == prefix;
        int sel   = above || (tie && trank < need);
        if (tie) trank++;
        sel8[j] = sel;
        lsel += sel;
    }
    __syncthreads();
    part[t] = lsel;
    __syncthreads();
    if (t == 0) { int s = 0; for (int i = 0; i < 256; ++i) { scan[i] = s; s += part[i]; } }
    __syncthreads();
    int pos = scan[t];
    #pragma unroll
    for (int j = 0; j < 8; ++j) {
        int s_idx = seg * FULLSEG + i0 + j;       // token index within batch
        if (sel8[j]) {
            flat[b * 2048 + seg * KSEL + pos] = s_idx;
            selpos[(size_t)b * Sn + s_idx] = seg * KSEL + pos;
            pos++;
        } else {
            selpos[(size_t)b * Sn + s_idx] = -1;
        }
    }
}

// gather selected rows of x into bf16 compact batch (8192 x 1024)
__global__ void gather_kernel(const float* __restrict__ x, const int* __restrict__ flat,
                              u16* __restrict__ xsel) {
    int r = blockIdx.x;                 // 0..8191
    int b = r >> 11, j = r & 2047;
    int s = flat[b * 2048 + j];
    const float* src = x + ((size_t)b * Sn + s) * Dn;
    u16* dst = xsel + (size_t)r * Dn;
    for (int c = threadIdx.x; c < Dn; c += 256) dst[c] = f2bf(src[c]);
}

// ---------------------------------------------------------------------------
// Generic bf16 WMMA GEMM: C(MxN) = A(MxK,bf16) * BT(NxK,bf16)^T [+bias][relu]
// 128x128 tile per block (8 waves, each wave 16 rows x 128 cols), BK=32.
// Global->LDS staging uses async-to-LDS b128 with double buffering:
// issue tile kt+1 while WMMAs consume tile kt; s_wait_asynccnt 4 keeps the
// next tile's 4 per-thread copies in flight.
__global__ __launch_bounds__(256) void gemm_kernel(
        const u16* __restrict__ A, const u16* __restrict__ BT,
        const float* __restrict__ bias, float* __restrict__ Cf, u16* __restrict__ Cb,
        int M, int N, int K, int relu) {
    __shared__ u16 As[2][4096];
    __shared__ u16 Bs[2][4096];
    int t = threadIdx.x;
    int w = t >> 5, lane = t & 31, half = lane >> 4, l15 = lane & 15;
    int rowBase = blockIdx.y * 128;
    int colBase = blockIdx.x * 128;
    v8f acc[8];
    #pragma unroll
    for (int jt = 0; jt < 8; ++jt) acc[jt] = (v8f){0.f,0.f,0.f,0.f,0.f,0.f,0.f,0.f};
    int lr = t >> 1;                  // 0..127
    int lseg = (t & 1) * 16;          // 0 / 16
    const u16* Ag = A  + (size_t)(rowBase + lr) * K + lseg;
    const u16* Bg = BT + (size_t)(colBase + lr) * K + lseg;
    unsigned ldsA0 = lds_addr(&As[0][lr * 32 + lseg]);
    unsigned ldsB0 = lds_addr(&Bs[0][lr * 32 + lseg]);
    int nk = K >> 5;
    // prologue: tile 0 -> buffer 0
    async_b128(ldsA0, Ag);
    async_b128(ldsB0, Bg);
    for (int kt = 0; kt < nk; ++kt) {
        int cur = kt & 1;
        if (kt + 1 < nk) {
            int nxt = cur ^ 1;
            async_b128(ldsA0 + nxt * 8192, Ag + (kt + 1) * 32);
            async_b128(ldsB0 + nxt * 8192, Bg + (kt + 1) * 32);
            asm volatile("s_wait_asynccnt 0x4" ::: "memory");   // tile kt landed
        } else {
            asm volatile("s_wait_asynccnt 0x0" ::: "memory");
        }
        __syncthreads();
        const u16* Ac = As[cur];
        const u16* Bc = Bs[cur];
        BF16Frag a;
        #pragma unroll
        for (int v = 0; v < 8; ++v) {
            int kk = (v < 4 ? 2 * v : 16 + 2 * (v - 4)) + half * 8;
            a.u[v] = *(const unsigned*)&Ac[(w * 16 + l15) * 32 + kk];
        }
        #pragma unroll
        for (int jt = 0; jt < 8; ++jt) {
            BF16Frag bb;
            #pragma unroll
            for (int v = 0; v < 8; ++v) {
                int kk = (v < 4 ? 2 * v : 16 + 2 * (v - 4)) + half * 8;
                bb.u[v] = *(const unsigned*)&Bc[(jt * 16 + l15) * 32 + kk];
            }
            acc[jt] = __builtin_amdgcn_wmma_f32_16x16x32_bf16(
                          false, a.v, false, bb.v, (short)0, acc[jt], false, false);
        }
        __syncthreads();   // buffer consumed before next overwrite
    }
    #pragma unroll
    for (int jt = 0; jt < 8; ++jt) {
        int c = colBase + jt * 16 + l15;
        float bval = bias ? bias[c] : 0.f;
        #pragma unroll
        for (int v = 0; v < 8; ++v) {
            int r = rowBase + w * 16 + half * 8 + v;
            float val = acc[jt][v] + bval;
            if (relu) val = fmaxf(val, 0.f);
            if (Cf) Cf[(size_t)r * N + c] = val;
            else    Cb[(size_t)r * N + c] = f2bf(val);
        }
    }
}

// pred = h(bf16, Mx256) . Wm2 + bm2  (one wave per token)
__global__ void pred_kernel(const u16* __restrict__ hbuf, const float* __restrict__ Wm2,
                            const float* __restrict__ bm2, float* __restrict__ pred) {
    int w = threadIdx.x >> 5, lane = threadIdx.x & 31;
    int tok = blockIdx.x * 8 + w;
    const u16* hr = hbuf + (size_t)tok * 256;
    float s = 0.f;
    for (int i = lane; i < 256; i += 32) s += bf2f(hr[i]) * Wm2[i];
    s = wave_reduce_add(s);
    if (lane == 0) pred[tok] = s + bm2[0];
}

// pack v (per-(b,h) 2048x64 block) -> vT bf16 (64x2048)
__global__ void packvT_kernel(const float* __restrict__ vf, u16* __restrict__ vT) {
    int bh = blockIdx.x;                                  // 0..31
    const float* src = vf + (size_t)bh * 131072;
    u16* dst = vT + (size_t)bh * 131072;
    for (int e = threadIdx.x; e < 131072; e += 256) {
        int tk = e >> 6, d = e & 63;
        dst[d * 2048 + tk] = f2bf(src[e]);
    }
}

// per (b,h,seg): KV = elu1(k)^T @ v (64x64) and zrow = sum_t elu1(k)[t][:]
__global__ void ktv_kernel(const float* __restrict__ kf, const float* __restrict__ vf,
                           float* __restrict__ KV, float* __restrict__ zrow) {
    int id = blockIdx.x;                    // 128 = b*8*4
    int n = id & 3, h = (id >> 2) & 7, b = id >> 5;
    size_t base = ((size_t)(b * 8 + h) * 2048 + n * SEGn) * 64;
    const float* kb = kf + base;
    const float* vb = vf + base;
    __shared__ float kc[4096], vc[4096];
    int t = threadIdx.x;
    int i = t >> 2, jb = (t & 3) << 4;
    float acc[16];
    #pragma unroll
    for (int j = 0; j < 16; ++j) acc[j] = 0.f;
    float zacc = 0.f;
    for (int c = 0; c < 8; ++c) {
        __syncthreads();
        for (int e = t; e < 4096; e += 256) { kc[e] = kb[c * 4096 + e]; vc[e] = vb[c * 4096 + e]; }
        __syncthreads();
        for (int tt = 0; tt < 64; ++tt) {
            float s = elu1(kc[tt * 64 + i]);
            #pragma unroll
            for (int j = 0; j < 16; ++j) acc[j] += s * vc[tt * 64 + jb + j];
            zacc += s;
        }
    }
    float* KVo = KV + (size_t)id * 4096;
    #pragma unroll
    for (int j = 0; j < 16; ++j) KVo[i * 64 + jb + j] = acc[j];
    if (jb == 0) zrow[id * 64 + i] = zacc;
}

// ---------------------------------------------------------------------------
// Fused segment attention: per (b,h,seg, 64-row q tile).
// WMMA QK^T -> softmax (LDS reductions) -> WMMA exp*V, combined with
// compressive-memory retrieval (sq@mem)/(sq@z) and sigmoid gate.
__global__ void attn_kernel(const float* __restrict__ qf, const u16* __restrict__ kbf,
                            const u16* __restrict__ vT, const float* __restrict__ KV,
                            const float* __restrict__ zrow, const float* __restrict__ betas,
                            u16* __restrict__ attb) {
    extern __shared__ char smem[];
    u16*   qbf_s   = (u16*)(smem);                  //  8 KB  (64x64 bf16 q)
    float* sq_s    = (float*)(smem + 8192);         // 16 KB  elu1(q)
    float* mem_s   = (float*)(smem + 24576);        // 16 KB  accumulated memory
    float* amem_s  = (float*)(smem + 40960);        // 16 KB  sq@mem
    float* zv_s    = (float*)(smem + 57344);        // 256 B
    float* denom_s = (float*)(smem + 57600);        // 256 B
    float* rowmax_s= (float*)(smem + 57856);        // 256 B
    float* rowsum_s= (float*)(smem + 58112);        // 256 B
    float* red_s   = (float*)(smem + 58368);        //  8 KB  64x32 partials
    u16*   probs_s = (u16*)(smem + 66560);          // 64 KB  64x512 bf16 exp-scores

    int id = blockIdx.x;
    int qt = id & 7, n = (id >> 3) & 3, h = (id >> 5) & 7, b = id >> 8;
    int t = threadIdx.x;
    int w = t >> 5, lane = t & 31, half = lane >> 4, l15 = lane & 15;
    int rb = w & 3, ch = w >> 2;

    size_t bh = (size_t)(b * 8 + h);
    const float* qblk = qf  + (bh * 2048 + n * SEGn + qt * 64) * 64;
    const u16*   kblk = kbf + (bh * 2048 + n * SEGn) * 64;
    const u16*   vTbh = vT  + bh * 131072 + n * SEGn;            // [dv]*2048 + kp
    u16* attp = attb + (size_t)b * 1048576 + (size_t)n * 262144 + (size_t)h * 32768 + qt * 4096;
    int bh4 = (b * 8 + h) * 4;

    // Phase A: load q tile; build bf16 copy + elu1 copy
    for (int e = t; e < 4096; e += 256) {
        float f = qblk[e];
        qbf_s[e] = f2bf(f);
        sq_s[e]  = elu1(f);
    }
    // Phase B: accumulate memory state from segments j < n
    for (int e = t; e < 4096; e += 256) {
        float s = 0.f;
        for (int j = 0; j < n; ++j) s += KV[(size_t)(bh4 + j) * 4096 + e];
        mem_s[e] = s;
    }
    if (t < 64) {
        float s = 1.f / 64.f;
        for (int j = 0; j < n; ++j) s += zrow[(bh4 + j) * 64 + t];
        zv_s[t] = s;
    }
    __syncthreads();
    // Phase C: amem = sq@mem, denom = sq@z
    {
        int row = t >> 2, jb = (t & 3) << 4;
        float a16[16];
        #pragma unroll
        for (int j = 0; j < 16; ++j) a16[j] = 0.f;
        float den = 0.f;
        for (int kk = 0; kk < 64; ++kk) {
            float s = sq_s[row * 64 + kk];
            const float* mrow = &mem_s[kk * 64 + jb];
            #pragma unroll
            for (int j = 0; j < 16; ++j) a16[j] += s * mrow[j];
            den += s * zv_s[kk];
        }
        #pragma unroll
        for (int j = 0; j < 16; ++j) amem_s[row * 64 + jb + j] = a16[j];
        if (jb == 0) denom_s[row] = den;
    }
    __syncthreads();

    // Phase D: scores = q @ k^T / 8  via WMMA; wave covers 16 rows x 256 keys
    v8f sc[16];
    #pragma unroll
    for (int jt = 0; jt < 16; ++jt) sc[jt] = (v8f){0.f,0.f,0.f,0.f,0.f,0.f,0.f,0.f};
    #pragma unroll
    for (int ks = 0; ks < 2; ++ks) {
        BF16Frag a;
        #pragma unroll
        for (int v = 0; v < 8; ++v) {
            int kk = ks * 32 + (v < 4 ? 2 * v : 16 + 2 * (v - 4)) + half * 8;
            a.u[v] = *(const unsigned*)&qbf_s[(rb * 16 + l15) * 64 + kk];
        }
        #pragma unroll
        for (int jt = 0; jt < 16; ++jt) {
            int tok = ch * 256 + jt * 16 + l15;
            BF16Frag bb;
            #pragma unroll
            for (int v = 0; v < 8; ++v) {
                int kk = ks * 32 + (v < 4 ? 2 * v : 16 + 2 * (v - 4)) + half * 8;
                bb.u[v] = *(const unsigned*)&kblk[tok * 64 + kk];
            }
            sc[jt] = __builtin_amdgcn_wmma_f32_16x16x32_bf16(
                         false, a.v, false, bb.v, (short)0, sc[jt], false, false);
        }
    }
    // softmax: row max
    float m8[8];
    #pragma unroll
    for (int v = 0; v < 8; ++v) m8[v] = -3.0e38f;
    #pragma unroll
    for (int jt = 0; jt < 16; ++jt)
        #pragma unroll
        for (int v = 0; v < 8; ++v) {
            float xv = sc[jt][v] * 0.125f;
            sc[jt][v] = xv;
            m8[v] = fmaxf(m8[v], xv);
        }
    #pragma unroll
    for (int v = 0; v < 8; ++v)
        red_s[(rb * 16 + half * 8 + v) * 32 + ch * 16 + l15] = m8[v];
    __syncthreads();
    if (t < 64) {
        float m = -3.0e38f;
        for (int c = 0; c < 32; ++c) m = fmaxf(m, red_s[t * 32 + c]);
        rowmax_s[t] = m;
    }
    __syncthreads();
    float s8[8];
    #pragma unroll
    for (int v = 0; v < 8; ++v) s8[v] = 0.f;
    #pragma unroll
    for (int jt = 0; jt < 16; ++jt)
        #pragma unroll
        for (int v = 0; v < 8; ++v) {
            int R = rb * 16 + half * 8 + v;
            float e = __expf(sc[jt][v] - rowmax_s[R]);
            s8[v] += e;
            probs_s[R * 512 + ch * 256 + jt * 16 + l15] = f2bf(e);
        }
    #pragma unroll
    for (int v = 0; v < 8; ++v)
        red_s[(rb * 16 + half * 8 + v) * 32 + ch * 16 + l15] = s8[v];
    __syncthreads();
    if (t < 64) {
        float s = 0.f;
        for (int c = 0; c < 32; ++c) s += red_s[t * 32 + c];
        rowsum_s[t] = s;
    }
    __syncthreads();

    // Phase E: att_dot_raw = exp @ V via WMMA; wave covers 16 rows x 32 dv cols
    v8f av[2];
    av[0] = (v8f){0.f,0.f,0.f,0.f,0.f,0.f,0.f,0.f};
    av[1] = (v8f){0.f,0.f,0.f,0.f,0.f,0.f,0.f,0.f};
    for (int kst = 0; kst < 16; ++kst) {
        BF16Frag a;
        #pragma unroll
        for (int v = 0; v < 8; ++v) {
            int kk = kst * 32 + (v < 4 ? 2 * v : 16 + 2 * (v - 4)) + half * 8;
            a.u[v] = *(const unsigned*)&probs_s[(rb * 16 + l15) * 512 + kk];
        }
        #pragma unroll
        for (int tt = 0; tt < 2; ++tt) {
            int dv = ch * 32 + tt * 16 + l15;
            BF16Frag bb;
            #pragma unroll
            for (int v = 0; v < 8; ++v) {
                int kk = kst * 32 + (v < 4 ? 2 * v : 16 + 2 * (v - 4)) + half * 8;
                bb.u[v] = *(const unsigned*)&vTbh[dv * 2048 + kk];
            }
            av[tt] = __builtin_amdgcn_wmma_f32_16x16x32_bf16(
                         false, a.v, false, bb.v, (short)0, av[tt], false, false);
        }
    }
    // Phase F: gate-combine and store bf16
    #pragma unroll
    for (int tt = 0; tt < 2; ++tt)
        #pragma unroll
        for (int v = 0; v < 8; ++v) {
            int R  = rb * 16 + half * 8 + v;
            int dv = ch * 32 + tt * 16 + l15;
            float g  = 1.f / (1.f + __expf(-betas[h * 64 + dv]));
            float am = amem_s[R * 64 + dv] / denom_s[R];
            float ad = av[tt][v] / rowsum_s[R];
            attp[R * 64 + dv] = f2bf(g * am + (1.f - g) * ad);
        }
}

// scatter-add residual + LayerNorm + sel_mask; one block per token
__global__ void ln_kernel(const float* __restrict__ x, const float* __restrict__ yf,
                          const int* __restrict__ selpos, const float* __restrict__ g,
                          const float* __restrict__ bb, float* __restrict__ out,
                          float* __restrict__ mask) {
    int tok = blockIdx.x, t = threadIdx.x;
    int b = tok >> 13;
    const float* xr = x + (size_t)tok * Dn;
    int sp = selpos[tok];
    const float* yr = (sp >= 0) ? (yf + ((size_t)b * 2048 + sp) * Dn) : nullptr;
    float v4[4];
    float psum = 0.f;
    #pragma unroll
    for (int j = 0; j < 4; ++j) {
        int c = t * 4 + j;
        float val = xr[c] + (sp >= 0 ? yr[c] : 0.f);
        v4[j] = val;
        psum += val;
    }
    __shared__ float red[256];
    red[t] = psum;
    __syncthreads();
    for (int off = 128; off; off >>= 1) { if (t < off) red[t] += red[t + off]; __syncthreads(); }
    float mu = red[0] * (1.f / Dn);
    __syncthreads();
    float pv = 0.f;
    #pragma unroll
    for (int j = 0; j < 4; ++j) { float d = v4[j] - mu; pv += d * d; }
    red[t] = pv;
    __syncthreads();
    for (int off = 128; off; off >>= 1) { if (t < off) red[t] += red[t + off]; __syncthreads(); }
    float rs = rsqrtf(red[0] * (1.f / Dn) + 1e-5f);
    #pragma unroll
    for (int j = 0; j < 4; ++j) {
        int c = t * 4 + j;
        out[(size_t)tok * Dn + c] = (v4[j] - mu) * rs * g[c] + bb[c];
    }
    if (t == 0) mask[tok] = (sp >= 0) ? 1.f : 0.f;
}

// ---------------------------------------------------------------------------
extern "C" void kernel_launch(void* const* d_in, const int* in_sizes, int n_in,
                              void* d_out, int out_size, void* d_ws, size_t ws_size,
                              hipStream_t stream) {
    const float* x    = (const float*)d_in[0];
    const float* Wq   = (const float*)d_in[1];
    const float* Wk   = (const float*)d_in[2];
    const float* Wv   = (const float*)d_in[3];
    const float* Wo   = (const float*)d_in[4];
    const float* betas= (const float*)d_in[5];
    const float* W1   = (const float*)d_in[6];
    const float* b1   = (const float*)d_in[7];
    const float* W2   = (const float*)d_in[8];
    const float* b2   = (const float*)d_in[9];
    const float* ln_g = (const float*)d_in[10];
    const float* ln_b = (const float*)d_in[11];
    const float* Wps  = (const float*)d_in[12];
    const float* bps  = (const float*)d_in[13];
    const float* Wm1  = (const float*)d_in[14];
    const float* bm1  = (const float*)d_in[15];
    const float* Wm2  = (const float*)d_in[16];
    const float* bm2  = (const float*)d_in[17];

    char* ws = (char*)d_ws;
    // workspace layout (bytes), 256-aligned blocks
    u16*   xbf    = (u16*)  (ws + 0);            // 33554432 bf16 (aliased by h1 later)
    u16*   xselbf = (u16*)  (ws + 67108864);     // 8192x1024 bf16
    u16*   WqT    = (u16*)  (ws + 83886080);
    u16*   WkT    = (u16*)  (ws + 84934656);
    u16*   WvT    = (u16*)  (ws + 85983232);
    u16*   WoT    = (u16*)  (ws + 87031808);
    u16*   W1T    = (u16*)  (ws + 88080384);     // 4096x1024
    u16*   W2T    = (u16*)  (ws + 96468992);     // 1024x4096
    u16*   Wm1T   = (u16*)  (ws + 104857600);    // 256x1024
    u16*   hbuf   = (u16*)  (ws + 105381888);    // 32768x256 bf16
    float* scoresb= (float*)(ws + 122159104);    // 32768
    int*   flatb  = (int*)  (ws + 122290176);    // 4x2048
    int*   selposb= (int*)  (ws + 122322944);    // 32768
    float* qfb    = (float*)(ws + 122454016);    // 8192x512 f32
    float* kfb    = (float*)(ws + 139231232);    // 8192x512 f32
    float* vfb    = (float*)(ws + 156008448);    // 8192x512 f32
    u16*   kbfb   = (u16*)  (ws + 172785664);    // 8192x512 bf16
    u16*   vTb    = (u16*)  (ws + 181174272);    // 8192x512 bf16 (transposed blocks)
    float* KVb    = (float*)(ws + 189562880);    // 128x4096 f32
    float* zrowb  = (float*)(ws + 191660032);    // 128x64 f32
    u16*   attb   = (u16*)  (ws + 191692800);    // 8192x512 bf16
    // aliases (non-overlapping lifetimes):
    u16*   ybf0   = (u16*)  (ws + 156008448);    // 8192x1024 bf16 (over vf)
    u16*   h1b    = (u16*)  (ws + 0);            // 8192x4096 bf16 (over xbf)
    float* yfb    = (float*)(ws + 122454016);    // 8192x1024 f32 (over qf+kf)

    float* out  = (float*)d_out;
    float* mask = out + (size_t)Bn * Sn * Dn;     // +33554432
    float* pred = mask + (size_t)Bn * Sn;         // +32768

    // 0) weight transpose/convert to bf16 (N x K layout for WMMA B-operand)
    tw_kernel<<<(1024 * 512 + 255) / 256, 256, 0, stream>>>(Wq, WqT, 1024, 512);
    tw_kernel<<<(1024 * 512 + 255) / 256, 256, 0, stream>>>(Wk, WkT, 1024, 512);
    tw_kernel<<<(1024 * 512 + 255) / 256, 256, 0, stream>>>(Wv, WvT, 1024, 512);
    tw_kernel<<<(512 * 1024 + 255) / 256, 256, 0, stream>>>(Wo, WoT, 512, 1024);
    tw_kernel<<<(1024 * 4096 + 255) / 256, 256, 0, stream>>>(W1, W1T, 1024, 4096);
    tw_kernel<<<(4096 * 1024 + 255) / 256, 256, 0, stream>>>(W2, W2T, 4096, 1024);
    tw_kernel<<<(1024 * 256 + 255) / 256, 256, 0, stream>>>(Wm1, Wm1T, 1024, 256);
    // 1) x -> bf16
    cvt_kernel<<<(33554432 / 4 + 255) / 256, 256, 0, stream>>>(x, xbf, 33554432);
    // 2) router scores + selection + gather
    scores_kernel<<<4096, 256, 0, stream>>>(x, Wps, bps, scoresb);
    select_kernel<<<16, 256, 0, stream>>>(scoresb, flatb, selposb);
    gather_kernel<<<8192, 256, 0, stream>>>(x, flatb, xselbf);
    // 3) aux prediction MLP (WMMA) + GEMV
    gemm_kernel<<<dim3(2, 256), 256, 0, stream>>>(xbf, Wm1T, bm1, nullptr, hbuf,
                                                  32768, 256, 1024, 1);
    pred_kernel<<<4096, 256, 0, stream>>>(hbuf, Wm2, bm2, pred);
    // 4) QKV projections (WMMA, fp32 out)
    gemm_kernel<<<dim3(4, 64), 256, 0, stream>>>(xselbf, WqT, nullptr, qfb, nullptr,
                                                 8192, 512, 1024, 0);
    gemm_kernel<<<dim3(4, 64), 256, 0, stream>>>(xselbf, WkT, nullptr, kfb, nullptr,
                                                 8192, 512, 1024, 0);
    gemm_kernel<<<dim3(4, 64), 256, 0, stream>>>(xselbf, WvT, nullptr, vfb, nullptr,
                                                 8192, 512, 1024, 0);
    // 5) attention prep: bf16 k, transposed bf16 v, linear-memory accumulators
    cvt_kernel<<<(4194304 / 4 + 255) / 256, 256, 0, stream>>>(kfb, kbfb, 4194304);
    packvT_kernel<<<32, 256, 0, stream>>>(vfb, vTb);
    ktv_kernel<<<128, 256, 0, stream>>>(kfb, vfb, KVb, zrowb);
    // 6) fused attention (WMMA QK^T + softmax + WMMA PV + compressive memory)
    attn_kernel<<<1024, 256, 132096, stream>>>(qfb, kbfb, vTb, KVb, zrowb, betas, attb);
    // 7) Wo + MLP (WMMA)
    gemm_kernel<<<dim3(8, 64), 256, 0, stream>>>(attb, WoT, nullptr, nullptr, ybf0,
                                                 8192, 1024, 512, 0);
    gemm_kernel<<<dim3(32, 64), 256, 0, stream>>>(ybf0, W1T, b1, nullptr, h1b,
                                                  8192, 4096, 1024, 1);
    gemm_kernel<<<dim3(8, 64), 256, 0, stream>>>(h1b, W2T, b2, yfb, nullptr,
                                                 8192, 1024, 4096, 0);
    // 8) scatter-add residual + LayerNorm + mask
    ln_kernel<<<32768, 256, 0, stream>>>(x, yfb, selposb, ln_g, ln_b, out, mask);
    (void)in_sizes; (void)n_in; (void)out_size; (void)ws_size;
}
